// VQ_12275016532437
// MI455X (gfx1250) — matrix operand order repaired and checked by reading, not compiled, not run on hardware
//
#include <hip/hip_runtime.h>

// ---------------------------------------------------------------------------
// VQ (vector quantization) for MI455X / gfx1250, wave32 + WMMA bf16.
//   x:      (16, 4096, 128) f32   -> q rows BN = 65536, D = 128
//   weight: (2048, 128) f32       -> M = 2048 codes
// Outputs (flat f32 in d_out): quantized_st (BN*D) | loss (1) | perp (1) | indices (BN)
// ---------------------------------------------------------------------------

typedef __attribute__((ext_vector_type(16))) __bf16 v16bf;
typedef __attribute__((ext_vector_type(8)))  float  v8f;

#define VQ_BN      65536
#define VQ_D       128
#define VQ_M       2048
#define VQ_RPB     32                 // q rows per block (2 WMMA row tiles)
#define VQ_NBLK    (VQ_BN / VQ_RPB)   // 2048 main blocks
#define VQ_QELEMS  (VQ_BN * VQ_D)     // 8388608

// ---------------------------------------------------------------------------
// Kernel 1: split codebook into bf16 hi/lo, exact f32 row norms, zero counts.
// ---------------------------------------------------------------------------
__global__ __launch_bounds__(128)
void vq_prep(const float* __restrict__ w, __bf16* __restrict__ w_hi,
             __bf16* __restrict__ w_lo, float* __restrict__ l2k,
             unsigned* __restrict__ counts)
{
    const int row = blockIdx.x;       // 0..M-1
    const int t   = threadIdx.x;      // 0..127
    const float v = w[row * VQ_D + t];
    const __bf16 h = (__bf16)v;
    const __bf16 l = (__bf16)(v - (float)h);
    w_hi[row * VQ_D + t] = h;
    w_lo[row * VQ_D + t] = l;

    __shared__ float s[128];
    s[t] = v * v;
    __syncthreads();
    for (int off = 64; off > 0; off >>= 1) {
        if (t < off) s[t] += s[t + off];
        __syncthreads();
    }
    if (t == 0) {
        l2k[row]    = s[0];
        counts[row] = 0u;
    }
}

// ---------------------------------------------------------------------------
// Kernel 2: per 32-row tile, full codebook scan with bf16 split-WMMA,
// argmin, gather + ST output, loss partial, histogram.
// ---------------------------------------------------------------------------
__global__ __launch_bounds__(256)
void vq_main(const float* __restrict__ x, const float* __restrict__ w,
             const __bf16* __restrict__ w_hi, const __bf16* __restrict__ w_lo,
             const float* __restrict__ l2k, unsigned* __restrict__ counts,
             float* __restrict__ loss_partial,
             float* __restrict__ out_q, float* __restrict__ out_idx)
{
    const int tile    = blockIdx.x;        // 0..VQ_NBLK-1
    const int tid     = threadIdx.x;       // 0..255 (8 waves)
    const int wave    = tid >> 5;
    const int lane    = tid & 31;
    const int half    = lane >> 4;         // 0: lanes 0-15, 1: lanes 16-31
    const int nsel    = lane & 15;
    const int rowbase = tile * VQ_RPB;

    // ---- Build WMMA A operands (two 16x128 row tiles, hi/lo bf16 split) ----
    // A k order used here (not AMD table order): lane holds row M=(lane&15); lanes<16 K=0..15+chunk,
    // lanes>=16 K=16..31+chunk (contiguous, matching B).
    v16bf a_hi[2][4], a_lo[2][4];
    #pragma unroll
    for (int rt = 0; rt < 2; ++rt) {
        const float* xr = x + (size_t)(rowbase + 16 * rt + nsel) * VQ_D;
        #pragma unroll
        for (int c = 0; c < 4; ++c) {
            const int k0 = 32 * c + 16 * half;
            const float4 p0 = *(const float4*)(xr + k0);
            const float4 p1 = *(const float4*)(xr + k0 + 4);
            const float4 p2 = *(const float4*)(xr + k0 + 8);
            const float4 p3 = *(const float4*)(xr + k0 + 12);
            const float lo8[8]  = {p0.x, p0.y, p0.z, p0.w, p1.x, p1.y, p1.z, p1.w};
            const float hi8[8]  = {p2.x, p2.y, p2.z, p2.w, p3.x, p3.y, p3.z, p3.w};
            #pragma unroll
            for (int e = 0; e < 8; ++e) {
                const __bf16 h0 = (__bf16)lo8[e];
                const __bf16 h1 = (__bf16)hi8[e];
                a_hi[rt][c][e]     = h0;
                a_lo[rt][c][e]     = (__bf16)(lo8[e] - (float)h0);
                a_hi[rt][c][8 + e] = h1;
                a_lo[rt][c][8 + e] = (__bf16)(hi8[e] - (float)h1);
            }
        }
    }

    // ---- Scan this wave's 256 codes (16 tiles of 16) ----
    float mval[2][8];
    int   midx[2][8];
    #pragma unroll
    for (int rt = 0; rt < 2; ++rt)
        #pragma unroll
        for (int r = 0; r < 8; ++r) { mval[rt][r] = 3.4e38f; midx[rt][r] = 0; }

    for (int t = 0; t < 16; ++t) {
        const int codebase = (wave * 16 + t) * 16;
        const int code     = codebase + nsel;   // this lane's B column / code id

        // B k order used here (same as A, not AMD table order): lane holds column N=(lane&15); lanes<16 K=0..15,
        // lanes>=16 K=16..31 of each 32-K chunk -> 16 contiguous bf16 per lane.
        v16bf b_hi[4], b_lo[4];
        const __bf16* bh = w_hi + (size_t)code * VQ_D + 16 * half;
        const __bf16* bl = w_lo + (size_t)code * VQ_D + 16 * half;
        #pragma unroll
        for (int c = 0; c < 4; ++c) {
            b_hi[c] = *(const v16bf*)(bh + 32 * c);
            b_lo[c] = *(const v16bf*)(bl + 32 * c);
        }

        const float lk = l2k[code];

        #pragma unroll
        for (int rt = 0; rt < 2; ++rt) {
            v8f acc0 = {};   // hi*hi
            v8f acc1 = {};   // hi*lo + lo*hi (error compensation)
            #pragma unroll
            for (int c = 0; c < 4; ++c) {
                acc0 = __builtin_amdgcn_wmma_f32_16x16x32_bf16(
                    false, a_hi[rt][c], false, b_hi[c], (short)0, acc0, false, false);
                acc1 = __builtin_amdgcn_wmma_f32_16x16x32_bf16(
                    false, a_hi[rt][c], false, b_lo[c], (short)0, acc1, false, false);
                acc1 = __builtin_amdgcn_wmma_f32_16x16x32_bf16(
                    false, a_lo[rt][c], false, b_hi[c], (short)0, acc1, false, false);
            }
            // C/D layout: VGPR r, lane -> (M = r + 8*half, N = lane&15).
            #pragma unroll
            for (int r = 0; r < 8; ++r) {
                const float dist = lk - 2.0f * (acc0[r] + acc1[r]);
                if (dist < mval[rt][r]) { mval[rt][r] = dist; midx[rt][r] = code; }
            }
        }
    }

    // ---- Cross-lane argmin within each 16-lane group (rows stay fixed) ----
    #pragma unroll
    for (int m = 1; m < 16; m <<= 1) {
        #pragma unroll
        for (int rt = 0; rt < 2; ++rt)
            #pragma unroll
            for (int r = 0; r < 8; ++r) {
                const float ov = __shfl_xor(mval[rt][r], m, 32);
                const int   oi = __shfl_xor(midx[rt][r], m, 32);
                if (ov < mval[rt][r] || (ov == mval[rt][r] && oi < midx[rt][r])) {
                    mval[rt][r] = ov; midx[rt][r] = oi;
                }
            }
    }

    // ---- Cross-wave argmin via LDS ----
    __shared__ float s_val[8][VQ_RPB];
    __shared__ int   s_idx[8][VQ_RPB];
    __shared__ int   s_final[VQ_RPB];
    if (nsel == 0) {  // lanes 0 and 16: rows 8*half..8*half+7 of each row tile
        #pragma unroll
        for (int rt = 0; rt < 2; ++rt)
            #pragma unroll
            for (int r = 0; r < 8; ++r) {
                s_val[wave][16 * rt + 8 * half + r] = mval[rt][r];
                s_idx[wave][16 * rt + 8 * half + r] = midx[rt][r];
            }
    }
    __syncthreads();

    if (tid < VQ_RPB) {
        float bv = s_val[0][tid];
        int   bi = s_idx[0][tid];
        #pragma unroll
        for (int wv = 1; wv < 8; ++wv) {
            const float v2 = s_val[wv][tid];
            const int   i2 = s_idx[wv][tid];
            if (v2 < bv || (v2 == bv && i2 < bi)) { bv = v2; bi = i2; }
        }
        s_final[tid] = bi;
        out_idx[rowbase + tid] = (float)bi;          // indices region (f32)
        atomicAdd(&counts[bi], 1u);                  // exact int histogram
    }
    __syncthreads();

    // ---- Gather z = weight[idx], write quantized_st (== z), loss partial ----
    float sq = 0.0f;
    #pragma unroll
    for (int rt = 0; rt < 2; ++rt) {
        const int r    = rt * 16 + (tid >> 4);        // row in tile
        const int c0   = (tid & 15) * 8;              // 8 cols per thread
        const int code = s_final[r];
        const float* wr  = w + (size_t)code * VQ_D + c0;
        const float* xr2 = x + (size_t)(rowbase + r) * VQ_D + c0;
        float*       oq  = out_q + (size_t)(rowbase + r) * VQ_D + c0;
        #pragma unroll
        for (int e = 0; e < 8; ++e) {
            const float z = wr[e];
            const float d = z - xr2[e];
            oq[e] = z;
            sq += d * d;
        }
    }
    __shared__ float s_loss[256];
    s_loss[tid] = sq;
    __syncthreads();
    for (int off = 128; off > 0; off >>= 1) {
        if (tid < off) s_loss[tid] += s_loss[tid + off];
        __syncthreads();
    }
    if (tid == 0) loss_partial[tile] = s_loss[0];
}

// ---------------------------------------------------------------------------
// Kernel 3: deterministic final reductions -> loss, perplexity.
// ---------------------------------------------------------------------------
__global__ __launch_bounds__(256)
void vq_finalize(const float* __restrict__ loss_partial,
                 const unsigned* __restrict__ counts,
                 float* __restrict__ out_loss, float* __restrict__ out_perp)
{
    __shared__ float s[256];
    const int tid = threadIdx.x;

    float acc = 0.0f;
    for (int i = tid; i < VQ_NBLK; i += 256) acc += loss_partial[i];
    s[tid] = acc;
    __syncthreads();
    for (int off = 128; off > 0; off >>= 1) {
        if (tid < off) s[tid] += s[tid + off];
        __syncthreads();
    }
    const float loss = s[0] / (float)VQ_QELEMS;
    __syncthreads();

    float ent = 0.0f;
    for (int i = tid; i < VQ_M; i += 256) {
        const float p = (float)counts[i] / (float)VQ_BN;
        ent += p * logf(p + 1e-10f);
    }
    s[tid] = ent;
    __syncthreads();
    for (int off = 128; off > 0; off >>= 1) {
        if (tid < off) s[tid] += s[tid + off];
        __syncthreads();
    }
    if (tid == 0) {
        *out_loss = loss;
        *out_perp = expf(-s[0]);
    }
}

// ---------------------------------------------------------------------------
extern "C" void kernel_launch(void* const* d_in, const int* in_sizes, int n_in,
                              void* d_out, int out_size, void* d_ws, size_t ws_size,
                              hipStream_t stream)
{
    (void)in_sizes; (void)n_in; (void)out_size; (void)ws_size;
    const float* x = (const float*)d_in[0];
    const float* w = (const float*)d_in[1];

    // Workspace layout (all offsets 256B-aligned):
    char* ws = (char*)d_ws;
    __bf16*   w_hi         = (__bf16*)(ws);                  // 524288 B
    __bf16*   w_lo         = (__bf16*)(ws + 524288);         // 524288 B
    float*    l2k          = (float*) (ws + 1048576);        //   8192 B
    unsigned* counts       = (unsigned*)(ws + 1056768);      //   8192 B
    float*    loss_partial = (float*) (ws + 1064960);        //   8192 B (VQ_NBLK)

    // d_out regions (f32): quantized | loss | perp | indices
    float* out       = (float*)d_out;
    float* out_q     = out;
    float* out_loss  = out + VQ_QELEMS;
    float* out_perp  = out + VQ_QELEMS + 1;
    float* out_idx   = out + VQ_QELEMS + 2;

    vq_prep<<<VQ_M, 128, 0, stream>>>(w, w_hi, w_lo, l2k, counts);
    vq_main<<<VQ_NBLK, 256, 0, stream>>>(x, w, w_hi, w_lo, l2k, counts,
                                         loss_partial, out_q, out_idx);
    vq_finalize<<<1, 256, 0, stream>>>(loss_partial, counts, out_loss, out_perp);
}
